// TransformerBlock_53395033424081
// MI455X (gfx1250) — matrix operand order repaired and checked by reading, not compiled, not run on hardware
//
#include <hip/hip_runtime.h>
#include <math.h>

// ---------------------------------------------------------------------------
// Types
// ---------------------------------------------------------------------------
typedef _Float16 half_t;
typedef __attribute__((ext_vector_type(16))) _Float16 v16h;
typedef __attribute__((ext_vector_type(8)))  _Float16 v8h;
typedef __attribute__((ext_vector_type(8)))  float    v8f;

union V16 { v16h v; v8h h[2]; };

__device__ __forceinline__ v8f wmma_f16(v16h a, v16h b, v8f c) {
    // D = A(16x32 f16) * B(32x16 f16) + C(16x16 f32)
    return __builtin_amdgcn_wmma_f32_16x16x32_f16(
        /*neg_a=*/false, a, /*neg_b=*/false, b,
        /*c_mod=*/(short)0, c, /*reuse_a=*/false, /*reuse_b=*/false);
}

// CDNA5 async global->LDS copy of 32 contiguous bytes (2 x b128).
// VDST = LDS byte address (low 32 bits of the generic shared pointer;
// flat aperture keeps the LDS offset in addr[31:0]).  INST_OFFSET is added
// to both the LDS and the memory address (ISA 08 section 4.4).
__device__ __forceinline__ void async_copy32B(const half_t* g, half_t* l) {
    unsigned la = (unsigned)(size_t)l;
    asm volatile("global_load_async_to_lds_b128 %0, %1, off"
                 :: "v"(la), "v"(g) : "memory");
    asm volatile("global_load_async_to_lds_b128 %0, %1, off offset:16"
                 :: "v"(la), "v"(g) : "memory");
}
__device__ __forceinline__ void wait_async0() {
    asm volatile("s_wait_asynccnt 0x0" ::: "memory");
}

// Problem constants
#define SEQ   2048
#define DIM   2048
#define HEADS 16
#define HD    128
#define HID   8192
#define LTCN  256

// ---------------------------------------------------------------------------
// Weight transpose + f32 -> f16 convert:  W[R][C] (row-major) -> Wt[C][R] f16
// ---------------------------------------------------------------------------
__global__ __launch_bounds__(256) void transpose_cvt_f32_to_f16(
    const float* __restrict__ W, half_t* __restrict__ Wt, int R, int C) {
    __shared__ float t[32][33];
    int c0 = blockIdx.x * 32, r0 = blockIdx.y * 32;
    int tx = threadIdx.x, ty = threadIdx.y;       // block (32, 8)
#pragma unroll
    for (int j = 0; j < 32; j += 8)
        t[ty + j][tx] = W[(size_t)(r0 + ty + j) * C + (c0 + tx)];
    __syncthreads();
#pragma unroll
    for (int j = 0; j < 32; j += 8)
        Wt[(size_t)(c0 + ty + j) * R + (r0 + tx)] = (half_t)t[tx][ty + j];
}

// f16 transpose: in[R][C] -> out[C][R]
__global__ __launch_bounds__(256) void transpose_f16(
    const half_t* __restrict__ in, half_t* __restrict__ out, int R, int C) {
    __shared__ half_t t[32][33];
    int c0 = blockIdx.x * 32, r0 = blockIdx.y * 32;
    int tx = threadIdx.x, ty = threadIdx.y;
#pragma unroll
    for (int j = 0; j < 32; j += 8)
        t[ty + j][tx] = in[(size_t)(r0 + ty + j) * C + (c0 + tx)];
    __syncthreads();
#pragma unroll
    for (int j = 0; j < 32; j += 8)
        out[(size_t)(c0 + ty + j) * R + (r0 + tx)] = t[tx][ty + j];
}

// ---------------------------------------------------------------------------
// RMSNorm (fp32 in, f16 out): one block per row
// ---------------------------------------------------------------------------
__global__ __launch_bounds__(256) void rmsnorm_f16(
    const float* __restrict__ x, const float* __restrict__ w,
    half_t* __restrict__ out, int D) {
    int row = blockIdx.x;
    const float* xr = x + (size_t)row * D;
    float ss = 0.0f;
    for (int j = threadIdx.x; j < D; j += 256) { float v = xr[j]; ss += v * v; }
    __shared__ float red[256];
    red[threadIdx.x] = ss;
    __syncthreads();
    for (int s = 128; s > 0; s >>= 1) {
        if (threadIdx.x < s) red[threadIdx.x] += red[threadIdx.x + s];
        __syncthreads();
    }
    float scale = rsqrtf(red[0] / (float)D + 1.1920929e-07f);
    half_t* o = out + (size_t)row * D;
    for (int j = threadIdx.x; j < D; j += 256)
        o[j] = (half_t)(xr[j] * scale * w[j]);
}

// ---------------------------------------------------------------------------
// RoPE on f16 spikes: grid (SEQ, HEADS), block 64
// ---------------------------------------------------------------------------
__global__ __launch_bounds__(64) void rope_f16(
    const half_t* __restrict__ in, half_t* __restrict__ out) {
    int j = threadIdx.x;                 // 0..63
    int s = blockIdx.x;
    int hh = blockIdx.y;
    float ang = (float)s * __expf(-((float)(2 * j) / 128.0f) * 9.2103403719762f);
    float sn, cs;
    __sincosf(ang, &sn, &cs);
    size_t base = (size_t)s * (HEADS * HD) + (size_t)hh * HD;
    float x1 = (float)in[base + j];
    float x2 = (float)in[base + 64 + j];
    out[base + j]      = (half_t)(x1 * cs - x2 * sn);
    out[base + 64 + j] = (half_t)(x2 * cs + x1 * sn);
}

// ---------------------------------------------------------------------------
// Generic WMMA GEMM: C[M,N] = A[M,K](f16,row-major) * Bt[N,K](f16) + epilogue
// block tile 128x128, BK=32, double-buffered async global->LDS pipeline.
// 8 waves, wave tile 32x64 (2x4 fragments).
// ---------------------------------------------------------------------------
enum { EPI_PLAIN = 0, EPI_SPIKE = 1, EPI_RES = 2, EPI_SILU = 3, EPI_ENH = 4 };

template <int EPI>
__global__ __launch_bounds__(256) void gemm_f16(
    const half_t* __restrict__ A, const half_t* __restrict__ B,
    const half_t* __restrict__ B2, const float* __restrict__ res,
    float* __restrict__ outF, half_t* __restrict__ outH,
    int M, int N, int K) {
    constexpr bool DUAL = (EPI == EPI_SILU);
    constexpr int BK  = 32;
    constexpr int LDT = 40;                       // padded row stride (halves)
    __shared__ half_t sA[2][128 * LDT];
    __shared__ half_t sB[2][128 * LDT];
    __shared__ half_t sB2[DUAL ? 2 : 1][DUAL ? 128 * LDT : 16];

    const int tid  = threadIdx.x;
    const int lane = tid & 31;
    const int wave = tid >> 5;
    const int wm = wave >> 1;                     // 0..3
    const int wn = wave & 1;                      // 0..1
    const int blockN = blockIdx.x * 128;
    const int blockM = blockIdx.y * 128;

    v8f acc[2][4];
    v8f acc2[DUAL ? 2 : 1][DUAL ? 4 : 1];
#pragma unroll
    for (int i = 0; i < 2; ++i)
#pragma unroll
        for (int j = 0; j < 4; ++j) {
            acc[i][j] = v8f{};
            if constexpr (DUAL) acc2[i][j] = v8f{};
        }

    // tile fill: each thread moves 16 contiguous halves (32B) per matrix
    const int lrow = tid >> 1;                    // 0..127
    const int lcol = (tid & 1) * 16;              // 0 or 16 (halves)
    const half_t* gA = A + (size_t)(blockM + lrow) * K + lcol;
    const half_t* gB = B + (size_t)(blockN + lrow) * K + lcol;
    const half_t* gB2 = nullptr;
    if constexpr (DUAL) gB2 = B2 + (size_t)(blockN + lrow) * K + lcol;
    const int ldst = lrow * LDT + lcol;

    const int nT = K / BK;
    // prefill tile 0
    async_copy32B(gA, &sA[0][ldst]);
    async_copy32B(gB, &sB[0][ldst]);
    if constexpr (DUAL) async_copy32B(gB2, &sB2[0][ldst]);
    wait_async0();
    __syncthreads();

    for (int t = 0; t < nT; ++t) {
        const int cur = t & 1;
        if (t + 1 < nT) {                         // async fill of next tile
            const size_t ko = (size_t)(t + 1) * BK;
            async_copy32B(gA + ko, &sA[cur ^ 1][ldst]);
            async_copy32B(gB + ko, &sB[cur ^ 1][ldst]);
            if constexpr (DUAL) async_copy32B(gB2 + ko, &sB2[cur ^ 1][ldst]);
        }
        if (t + 4 < nT) {                         // keep L2->WGP path warm
            __builtin_prefetch(gA + (size_t)(t + 4) * BK, 0, 1);
            __builtin_prefetch(gB + (size_t)(t + 4) * BK, 0, 1);
        }

        // ---- one K=32 WMMA step on the current tile ----
        V16 af[2];
        const int abase = (lane & 16) ? 16 : 0;
#pragma unroll
        for (int fm = 0; fm < 2; ++fm) {
            const half_t* ap = &sA[cur][(wm * 32 + fm * 16 + (lane & 15)) * LDT + abase];
            af[fm].h[0] = *(const v8h*)ap;
            af[fm].h[1] = *(const v8h*)(ap + 8);
        }
        const int bbase = (lane & 16) ? 16 : 0;
#pragma unroll
        for (int fn = 0; fn < 4; ++fn) {
            V16 bf;
            const half_t* bp = &sB[cur][(wn * 64 + fn * 16 + (lane & 15)) * LDT + bbase];
            bf.h[0] = *(const v8h*)bp;
            bf.h[1] = *(const v8h*)(bp + 8);
            acc[0][fn] = wmma_f16(af[0].v, bf.v, acc[0][fn]);
            acc[1][fn] = wmma_f16(af[1].v, bf.v, acc[1][fn]);
            if constexpr (DUAL) {
                V16 bg;
                const half_t* bp2 = &sB2[cur][(wn * 64 + fn * 16 + (lane & 15)) * LDT + bbase];
                bg.h[0] = *(const v8h*)bp2;
                bg.h[1] = *(const v8h*)(bp2 + 8);
                acc2[0][fn] = wmma_f16(af[0].v, bg.v, acc2[0][fn]);
                acc2[1][fn] = wmma_f16(af[1].v, bg.v, acc2[1][fn]);
            }
        }
        wait_async0();                            // own async fills landed
        __syncthreads();                          // all waves' fills visible
    }

    // ---- epilogue ----
#pragma unroll
    for (int fm = 0; fm < 2; ++fm)
#pragma unroll
        for (int fn = 0; fn < 4; ++fn)
#pragma unroll
            for (int i = 0; i < 8; ++i) {
                int m = wm * 32 + fm * 16 + i + ((lane & 16) ? 8 : 0);
                int n = wn * 64 + fn * 16 + (lane & 15);
                size_t idx = (size_t)(blockM + m) * N + (blockN + n);
                float v = acc[fm][fn][i];
                if constexpr (EPI == EPI_PLAIN) {
                    outF[idx] = v;
                } else if constexpr (EPI == EPI_SPIKE) {
                    outH[idx] = (half_t)((v > 1.0f) ? 1.0f : 0.0f);
                } else if constexpr (EPI == EPI_RES) {
                    outF[idx] = res[idx] + v;
                } else if constexpr (EPI == EPI_SILU) {
                    float u = acc2[fm][fn][i];
                    float s = (v / (1.0f + __expf(-v))) * u;    // silu(g)*u
                    outF[idx] = s;
                    outH[idx] = (half_t)s;
                } else if constexpr (EPI == EPI_ENH) {
                    outH[idx] = (half_t)(res[idx] + v);
                }
            }
}

// ---------------------------------------------------------------------------
// Flash attention: one wave32 per (16-query tile, head). Online softmax.
// rq/rk: [SEQ][HEADS*HD] f16 ; svT: [HEADS*HD][SEQ] f16 ; out: [SEQ][HEADS*HD]
// ---------------------------------------------------------------------------
__global__ __launch_bounds__(32) void flash_attn(
    const half_t* __restrict__ rq, const half_t* __restrict__ rk,
    const half_t* __restrict__ svT, half_t* __restrict__ out) {
    const int lane = threadIdx.x;
    const int head = blockIdx.y;
    const int q0 = blockIdx.x * 16;
    const float scale = 0.08838834764831845f;     // HD^-0.5
    __shared__ half_t sP[16 * 40];

    // Q fragments (A layout): 4 chunks of K=32 over HD=128
    V16 qf[4];
    {
        const int m = q0 + (lane & 15);
        const half_t* base = rq + (size_t)m * (HEADS * HD) + head * HD;
        const int koff = (lane & 16) ? 16 : 0;
#pragma unroll
        for (int kk = 0; kk < 4; ++kk) {
            const half_t* p = base + kk * 32 + koff;
            qf[kk].h[0] = *(const v8h*)p;
            qf[kk].h[1] = *(const v8h*)(p + 8);
        }
    }

    v8f o[8];
#pragma unroll
    for (int f = 0; f < 8; ++f) o[f] = v8f{};
    float mst[8], lst[8];
#pragma unroll
    for (int i = 0; i < 8; ++i) { mst[i] = -1e30f; lst[i] = 0.0f; }

    const int nChunks = (q0 + 16 + 31) / 32;
    for (int c = 0; c < nChunks; ++c) {
        const int kbase = c * 32;
        if (c + 1 < nChunks) {                    // prefetch next key chunk
            __builtin_prefetch(
                rk + (size_t)(kbase + 32 + (lane & 15)) * (HEADS * HD) + head * HD, 0, 1);
            __builtin_prefetch(
                svT + (size_t)(head * HD + (lane & 15)) * SEQ + kbase + 32, 0, 1);
        }
        v8f s0 = v8f{}, s1 = v8f{};
        // S = rq * rk^T  (B fragment: N=key, K=hd; contiguous along hd)
#pragma unroll
        for (int kk = 0; kk < 4; ++kk) {
            const int hb = kk * 32 + ((lane & 16) ? 16 : 0);
            V16 b0, b1;
            const half_t* p0 = rk + (size_t)(kbase + (lane & 15)) * (HEADS * HD) + head * HD + hb;
            b0.h[0] = *(const v8h*)p0;
            b0.h[1] = *(const v8h*)(p0 + 8);
            const half_t* p1 = rk + (size_t)(kbase + 16 + (lane & 15)) * (HEADS * HD) + head * HD + hb;
            b1.h[0] = *(const v8h*)p1;
            b1.h[1] = *(const v8h*)(p1 + 8);
            s0 = wmma_f16(qf[kk].v, b0.v, s0);
            s1 = wmma_f16(qf[kk].v, b1.v, s1);
        }
        // causal mask + scale + row max (rows live on (vgpr, lane-half))
        float pmax[8];
#pragma unroll
        for (int i = 0; i < 8; ++i) {
            int m = q0 + i + ((lane & 16) ? 8 : 0);
            int k0i = kbase + (lane & 15);
            float a = (k0i <= m) ? s0[i] * scale : -1e30f;
            float b = (k0i + 16 <= m) ? s1[i] * scale : -1e30f;
            s0[i] = a; s1[i] = b;
            float v = fmaxf(a, b);
#pragma unroll
            for (int off = 1; off < 16; off <<= 1)
                v = fmaxf(v, __shfl_xor(v, off, 32));
            pmax[i] = v;
        }
#pragma unroll
        for (int i = 0; i < 8; ++i) {
            float newm = fmaxf(mst[i], pmax[i]);
            float corr = __expf(mst[i] - newm);
            mst[i] = newm;
            float p0 = __expf(s0[i] - newm);
            float p1 = __expf(s1[i] - newm);
            s0[i] = p0; s1[i] = p1;
            float rs = p0 + p1;
#pragma unroll
            for (int off = 1; off < 16; off <<= 1)
                rs += __shfl_xor(rs, off, 32);
            lst[i] = lst[i] * corr + rs;
#pragma unroll
            for (int f = 0; f < 8; ++f) o[f][i] *= corr;
        }
        // spin P (C layout) through LDS into A layout
        __syncthreads();
#pragma unroll
        for (int i = 0; i < 8; ++i) {
            int m = i + ((lane & 16) ? 8 : 0);
            sP[m * 40 + (lane & 15)]      = (half_t)s0[i];
            sP[m * 40 + 16 + (lane & 15)] = (half_t)s1[i];
        }
        __syncthreads();
        V16 pf;
        {
            const int base = (lane & 16) ? 16 : 0;
            const half_t* pp = sP + (lane & 15) * 40 + base;
            pf.h[0] = *(const v8h*)pp;
            pf.h[1] = *(const v8h*)(pp + 8);
        }
        // O += P * V   (V B-frag from svT: N=hd, K=key contiguous)
        const int kb = kbase + ((lane & 16) ? 16 : 0);
#pragma unroll
        for (int f = 0; f < 8; ++f) {
            const half_t* vp = svT + (size_t)(head * HD + f * 16 + (lane & 15)) * SEQ + kb;
            V16 bv;
            bv.h[0] = *(const v8h*)vp;
            bv.h[1] = *(const v8h*)(vp + 8);
            o[f] = wmma_f16(pf.v, bv.v, o[f]);
        }
    }
    // finalize
    float rinv[8];
#pragma unroll
    for (int i = 0; i < 8; ++i) rinv[i] = 1.0f / lst[i];
#pragma unroll
    for (int f = 0; f < 8; ++f)
#pragma unroll
        for (int i = 0; i < 8; ++i) {
            int m = q0 + i + ((lane & 16) ? 8 : 0);
            out[(size_t)m * (HEADS * HD) + head * HD + f * 16 + (lane & 15)] =
                (half_t)(o[f][i] * rinv[i]);
        }
}

// ---------------------------------------------------------------------------
// LTC cell elementwise (h = 0 simplification):
// l = tanh( (g*tanh(ic)+ (1-g)*tanh(hb)) / (tau + 1e-8) )
// ---------------------------------------------------------------------------
__global__ __launch_bounds__(256) void ltc_elem(
    const float* __restrict__ t0, const float* __restrict__ g0,
    const float* __restrict__ i0, const float* __restrict__ tb,
    const float* __restrict__ gb, const float* __restrict__ ib,
    const float* __restrict__ hb, half_t* __restrict__ out) {
    int idx = blockIdx.x * 256 + threadIdx.x;
    int n = idx & (LTCN - 1);
    float tau = 0.1f + 9.9f / (1.0f + __expf(-(t0[idx] + tb[n])));
    float g = 1.0f / (1.0f + __expf(-(g0[idx] + gb[n])));
    float ic = tanhf(i0[idx] + ib[n]);
    float hc = tanhf(hb[n]);
    float tgt = g * ic + (1.0f - g) * hc;
    out[idx] = (half_t)tanhf(tgt / (tau + 1e-8f));
}

// ---------------------------------------------------------------------------
// Launcher
// ---------------------------------------------------------------------------
extern "C" void kernel_launch(void* const* d_in, const int* in_sizes, int n_in,
                              void* d_out, int out_size, void* d_ws, size_t ws_size,
                              hipStream_t stream) {
    (void)in_sizes; (void)n_in; (void)out_size; (void)ws_size;
    const float* x        = (const float*)d_in[0];
    const float* attn_nw  = (const float*)d_in[1];
    const float* ffn_nw   = (const float*)d_in[2];
    const float* q_w      = (const float*)d_in[3];
    const float* k_w      = (const float*)d_in[4];
    const float* v_w      = (const float*)d_in[5];
    const float* o_w      = (const float*)d_in[6];
    const float* gate_w   = (const float*)d_in[7];
    const float* up_w     = (const float*)d_in[8];
    const float* down_w   = (const float*)d_in[9];
    const float* proj_w   = (const float*)d_in[10];
    const float* c0_tau_w = (const float*)d_in[11];
    const float* c0_tau_b = (const float*)d_in[12];
    const float* c0_gate_w= (const float*)d_in[13];
    const float* c0_gate_b= (const float*)d_in[14];
    const float* c0_in_w  = (const float*)d_in[15];
    const float* c0_in_b  = (const float*)d_in[16];
    const float* c0_hid_b = (const float*)d_in[18];
    const float* c1_tau_w = (const float*)d_in[19];
    const float* c1_tau_b = (const float*)d_in[20];
    const float* c1_gate_w= (const float*)d_in[21];
    const float* c1_gate_b= (const float*)d_in[22];
    const float* c1_in_w  = (const float*)d_in[23];
    const float* c1_in_b  = (const float*)d_in[24];
    const float* c1_hid_b = (const float*)d_in[26];
    float* outp = (float*)d_out;

    char* wsb = (char*)d_ws;
    size_t off = 0;
    auto alloc = [&](size_t bytes) -> void* {
        void* p = wsb + off;
        off = (off + bytes + 255) & ~(size_t)255;
        return p;
    };
    const size_t SD = (size_t)SEQ * DIM;
    // f16 transposed weights
    half_t* qT  = (half_t*)alloc(SD * 2);
    half_t* kT  = (half_t*)alloc(SD * 2);
    half_t* vT  = (half_t*)alloc(SD * 2);
    half_t* oT  = (half_t*)alloc(SD * 2);
    half_t* gT  = (half_t*)alloc((size_t)HID * DIM * 2);
    half_t* uT  = (half_t*)alloc((size_t)HID * DIM * 2);
    half_t* dT  = (half_t*)alloc((size_t)DIM * HID * 2);
    half_t* pT  = (half_t*)alloc((size_t)HID * LTCN * 2);
    half_t* c0tT= (half_t*)alloc((size_t)LTCN * HID * 2);
    half_t* c0gT= (half_t*)alloc((size_t)LTCN * HID * 2);
    half_t* c0iT= (half_t*)alloc((size_t)LTCN * HID * 2);
    half_t* c1tT= (half_t*)alloc((size_t)LTCN * LTCN * 2);
    half_t* c1gT= (half_t*)alloc((size_t)LTCN * LTCN * 2);
    half_t* c1iT= (half_t*)alloc((size_t)LTCN * LTCN * 2);
    // activations
    half_t* xn16   = (half_t*)alloc(SD * 2);
    half_t* sq     = (half_t*)alloc(SD * 2);
    half_t* sk     = (half_t*)alloc(SD * 2);
    half_t* sv     = (half_t*)alloc(SD * 2);
    half_t* rqb    = (half_t*)alloc(SD * 2);
    half_t* rkb    = (half_t*)alloc(SD * 2);
    half_t* svT    = (half_t*)alloc(SD * 2);
    half_t* ao16   = (half_t*)alloc(SD * 2);
    float*  h32    = (float*) alloc(SD * 4);
    half_t* hn16   = (half_t*)alloc(SD * 2);
    float*  inter32= (float*) alloc((size_t)SEQ * HID * 4);
    half_t* inter16= (half_t*)alloc((size_t)SEQ * HID * 2);
    float*  t0     = (float*) alloc((size_t)SEQ * LTCN * 4);
    float*  g0     = (float*) alloc((size_t)SEQ * LTCN * 4);
    float*  i0     = (float*) alloc((size_t)SEQ * LTCN * 4);
    half_t* l0_16  = (half_t*)alloc((size_t)SEQ * LTCN * 2);
    float*  t1     = (float*) alloc((size_t)SEQ * LTCN * 4);
    float*  g1     = (float*) alloc((size_t)SEQ * LTCN * 4);
    float*  i1     = (float*) alloc((size_t)SEQ * LTCN * 4);
    half_t* l1_16  = (half_t*)alloc((size_t)SEQ * LTCN * 2);
    half_t* enh16  = (half_t*)alloc((size_t)SEQ * HID * 2);

    const dim3 tb(32, 8);
    // --- weight convert/transpose (f16, L2-resident thereafter) ---
    transpose_cvt_f32_to_f16<<<dim3(DIM/32, DIM/32), tb, 0, stream>>>(q_w,  qT,  DIM, DIM);
    transpose_cvt_f32_to_f16<<<dim3(DIM/32, DIM/32), tb, 0, stream>>>(k_w,  kT,  DIM, DIM);
    transpose_cvt_f32_to_f16<<<dim3(DIM/32, DIM/32), tb, 0, stream>>>(v_w,  vT,  DIM, DIM);
    transpose_cvt_f32_to_f16<<<dim3(DIM/32, DIM/32), tb, 0, stream>>>(o_w,  oT,  DIM, DIM);
    transpose_cvt_f32_to_f16<<<dim3(HID/32, DIM/32), tb, 0, stream>>>(gate_w, gT, DIM, HID);
    transpose_cvt_f32_to_f16<<<dim3(HID/32, DIM/32), tb, 0, stream>>>(up_w,   uT, DIM, HID);
    transpose_cvt_f32_to_f16<<<dim3(DIM/32, HID/32), tb, 0, stream>>>(down_w, dT, HID, DIM);
    transpose_cvt_f32_to_f16<<<dim3(HID/32, LTCN/32), tb, 0, stream>>>(proj_w, pT, LTCN, HID);
    transpose_cvt_f32_to_f16<<<dim3(LTCN/32, HID/32), tb, 0, stream>>>(c0_tau_w,  c0tT, HID, LTCN);
    transpose_cvt_f32_to_f16<<<dim3(LTCN/32, HID/32), tb, 0, stream>>>(c0_gate_w, c0gT, HID, LTCN);
    transpose_cvt_f32_to_f16<<<dim3(LTCN/32, HID/32), tb, 0, stream>>>(c0_in_w,   c0iT, HID, LTCN);
    transpose_cvt_f32_to_f16<<<dim3(LTCN/32, LTCN/32), tb, 0, stream>>>(c1_tau_w,  c1tT, LTCN, LTCN);
    transpose_cvt_f32_to_f16<<<dim3(LTCN/32, LTCN/32), tb, 0, stream>>>(c1_gate_w, c1gT, LTCN, LTCN);
    transpose_cvt_f32_to_f16<<<dim3(LTCN/32, LTCN/32), tb, 0, stream>>>(c1_in_w,   c1iT, LTCN, LTCN);

    // --- attention path ---
    rmsnorm_f16<<<SEQ, 256, 0, stream>>>(x, attn_nw, xn16, DIM);
    gemm_f16<EPI_SPIKE><<<dim3(DIM/128, SEQ/128), 256, 0, stream>>>(
        xn16, qT, nullptr, nullptr, nullptr, sq, SEQ, DIM, DIM);
    gemm_f16<EPI_SPIKE><<<dim3(DIM/128, SEQ/128), 256, 0, stream>>>(
        xn16, kT, nullptr, nullptr, nullptr, sk, SEQ, DIM, DIM);
    gemm_f16<EPI_SPIKE><<<dim3(DIM/128, SEQ/128), 256, 0, stream>>>(
        xn16, vT, nullptr, nullptr, nullptr, sv, SEQ, DIM, DIM);
    rope_f16<<<dim3(SEQ, HEADS), 64, 0, stream>>>(sq, rqb);
    rope_f16<<<dim3(SEQ, HEADS), 64, 0, stream>>>(sk, rkb);
    transpose_f16<<<dim3(DIM/32, SEQ/32), tb, 0, stream>>>(sv, svT, SEQ, DIM);
    flash_attn<<<dim3(SEQ/16, HEADS), 32, 0, stream>>>(rqb, rkb, svT, ao16);
    gemm_f16<EPI_RES><<<dim3(DIM/128, SEQ/128), 256, 0, stream>>>(
        ao16, oT, nullptr, x, h32, nullptr, SEQ, DIM, DIM);

    // --- FFN + LTC path ---
    rmsnorm_f16<<<SEQ, 256, 0, stream>>>(h32, ffn_nw, hn16, DIM);
    gemm_f16<EPI_SILU><<<dim3(HID/128, SEQ/128), 256, 0, stream>>>(
        hn16, gT, uT, nullptr, inter32, inter16, SEQ, HID, DIM);
    // LTC cell 0 (input = inter, hidden = 0)
    gemm_f16<EPI_PLAIN><<<dim3(LTCN/128, SEQ/128), 256, 0, stream>>>(
        inter16, c0tT, nullptr, nullptr, t0, nullptr, SEQ, LTCN, HID);
    gemm_f16<EPI_PLAIN><<<dim3(LTCN/128, SEQ/128), 256, 0, stream>>>(
        inter16, c0gT, nullptr, nullptr, g0, nullptr, SEQ, LTCN, HID);
    gemm_f16<EPI_PLAIN><<<dim3(LTCN/128, SEQ/128), 256, 0, stream>>>(
        inter16, c0iT, nullptr, nullptr, i0, nullptr, SEQ, LTCN, HID);
    ltc_elem<<<(SEQ * LTCN) / 256, 256, 0, stream>>>(
        t0, g0, i0, c0_tau_b, c0_gate_b, c0_in_b, c0_hid_b, l0_16);
    // LTC cell 1 (input = l0, hidden = 0)
    gemm_f16<EPI_PLAIN><<<dim3(LTCN/128, SEQ/128), 256, 0, stream>>>(
        l0_16, c1tT, nullptr, nullptr, t1, nullptr, SEQ, LTCN, LTCN);
    gemm_f16<EPI_PLAIN><<<dim3(LTCN/128, SEQ/128), 256, 0, stream>>>(
        l0_16, c1gT, nullptr, nullptr, g1, nullptr, SEQ, LTCN, LTCN);
    gemm_f16<EPI_PLAIN><<<dim3(LTCN/128, SEQ/128), 256, 0, stream>>>(
        l0_16, c1iT, nullptr, nullptr, i1, nullptr, SEQ, LTCN, LTCN);
    ltc_elem<<<(SEQ * LTCN) / 256, 256, 0, stream>>>(
        t1, g1, i1, c1_tau_b, c1_gate_b, c1_in_b, c1_hid_b, l1_16);
    // enhanced = inter + l1 @ ltc_proj_w (fused add, f16 out)
    gemm_f16<EPI_ENH><<<dim3(HID/128, SEQ/128), 256, 0, stream>>>(
        l1_16, pT, nullptr, inter32, nullptr, enh16, SEQ, HID, LTCN);
    // out = h + enhanced @ down_w
    gemm_f16<EPI_RES><<<dim3(DIM/128, SEQ/128), 256, 0, stream>>>(
        enh16, dT, nullptr, h32, outp, nullptr, SEQ, DIM, HID);
}
